// TemporalBlock_89395449299817
// MI455X (gfx1250) — compile-verified, matrix operand order checked by reading, not run on hardware
//
#include <hip/hip_runtime.h>
#include <hip/hip_bf16.h>
#include <math.h>

typedef __attribute__((ext_vector_type(16))) _Float16 v16h;
typedef __attribute__((ext_vector_type(8)))  _Float16 v8h;
typedef __attribute__((ext_vector_type(2)))  _Float16 v2h;
typedef __attribute__((ext_vector_type(8)))  float    v8f;

#define NB   4      // batch
#define NT   2048   // seq len
#define ND   512    // d_model
#define NH   8      // heads
#define NDH  64     // d_head
#define NWIN 10     // window
#define NFF  2048   // d_ff
#define LN_EPS 1e-5f

// ---------------------------------------------------------------------------
// Fragment-order packing. WMMA 16-bit A (16x32) layout per ISA 7.12.2:
//   lane L: row m = L%16, half-group hgrp = L/16
//   v16h element e: v = e/2, pos = e%2
//   k = (v<4) ? v*2+pos + 8*hgrp : 16 + (v-4)*2+pos + 8*hgrp
// B (32x16) mirrors A with n = L%16 instead of m.
// Packed storage: frag(tile, lane) contiguous 16 halfs; A tiles ordered
// (mt*KT + kt), B tiles ordered (nt*KT + kt), KT = K/32.
// ---------------------------------------------------------------------------
__device__ __forceinline__ float gelu_exact(float x) {
    return 0.5f * x * (1.0f + erff(x * 0.70710678118654752440f));
}

template<typename SrcT, bool GELU>
__global__ void pack_a_kernel(const SrcT* __restrict__ src, _Float16* __restrict__ dst,
                              int M, int K) {
    int id = blockIdx.x * blockDim.x + threadIdx.x;       // one dword (2 halfs)
    int total = (M * K) >> 1;
    if (id >= total) return;
    int KT   = K >> 5;
    int v    = id & 7;
    int lane = (id >> 3) & 31;
    int tile = id >> 8;
    int kt   = tile % KT;
    int mt   = tile / KT;
    int m    = mt * 16 + (lane & 15);
    int hg   = lane >> 4;
    int kb   = kt * 32 + ((v < 4) ? (v * 2 + 8 * hg) : (16 + (v - 4) * 2 + 8 * hg));
    float x0 = (float)src[(size_t)m * K + kb];
    float x1 = (float)src[(size_t)m * K + kb + 1];
    if (GELU) { x0 = gelu_exact(x0); x1 = gelu_exact(x1); }
    v2h o; o[0] = (_Float16)x0; o[1] = (_Float16)x1;
    ((v2h*)dst)[id] = o;
}

__global__ void pack_b_kernel(const float* __restrict__ src, _Float16* __restrict__ dst,
                              int K, int N) {   // src is [K, N] row-major
    int id = blockIdx.x * blockDim.x + threadIdx.x;
    int total = (K * N) >> 1;
    if (id >= total) return;
    int KT   = K >> 5;
    int v    = id & 7;
    int lane = (id >> 3) & 31;
    int tile = id >> 8;
    int kt   = tile % KT;
    int nt   = tile / KT;
    int n    = nt * 16 + (lane & 15);
    int hg   = lane >> 4;
    int kb   = kt * 32 + ((v < 4) ? (v * 2 + 8 * hg) : (16 + (v - 4) * 2 + 8 * hg));
    v2h o;
    o[0] = (_Float16)src[(size_t)kb * N + n];
    o[1] = (_Float16)src[(size_t)(kb + 1) * N + n];
    ((v2h*)dst)[id] = o;
}

__global__ void concat_bias_kernel(const float* bq, const float* bk, const float* bv,
                                   float* dst) {
    int i = blockIdx.x * blockDim.x + threadIdx.x;
    if (i < ND) { dst[i] = bq[i]; dst[i + ND] = bk[i]; dst[i + 2 * ND] = bv[i]; }
}

// ---------------------------------------------------------------------------
// WMMA GEMM: C[M,N] = A[M,K] * B[K,N] + bias[N].
// 256 threads = 8 waves; each wave computes a 64x32 tile (4x2 WMMA tiles).
// Register budget: 8 acc tiles (64 VGPR) + 4 A frags (32) + 2 B frags (16)
// + addressing  ->  fits without scratch spills (previous 4x4 tile spilled).
// A, B are fragment-packed f16; C is f16 or f32 (template).
// ---------------------------------------------------------------------------
template<bool F16OUT>
__global__ void __launch_bounds__(256)
gemm_wmma_kernel(const v16h* __restrict__ A, const v16h* __restrict__ B,
                 const float* __restrict__ bias, void* __restrict__ Cout,
                 int M, int N, int K) {
    const int lane = threadIdx.x & 31;
    const int wave = threadIdx.x >> 5;
    const int tn   = N >> 5;                       // 32-wide tiles along N
    const int wid  = blockIdx.x * 8 + wave;
    const int tile_n = wid % tn;
    const int tile_m = wid / tn;
    const int KT  = K >> 5;
    const int mt0 = tile_m * 4;
    const int nt0 = tile_n * 2;

    v8f acc[4][2];
    const v8f vzero = {0.f, 0.f, 0.f, 0.f, 0.f, 0.f, 0.f, 0.f};
#pragma unroll
    for (int i = 0; i < 4; ++i)
#pragma unroll
        for (int j = 0; j < 2; ++j) acc[i][j] = vzero;

    // Hoisted fragment base pointers: frag(i or j, kt) at base + i*KT*32 + kt*32.
    const v16h* __restrict__ Ab = A + ((size_t)mt0 * KT) * 32 + lane;
    const v16h* __restrict__ Bb = B + ((size_t)nt0 * KT) * 32 + lane;

    for (int kt = 0; kt < KT; ++kt) {
        v16h af[4], bf[2];
#pragma unroll
        for (int i = 0; i < 4; ++i)
            af[i] = Ab[((size_t)i * KT + kt) * 32];
#pragma unroll
        for (int j = 0; j < 2; ++j)
            bf[j] = Bb[((size_t)j * KT + kt) * 32];
#pragma unroll
        for (int i = 0; i < 4; ++i)
#pragma unroll
            for (int j = 0; j < 2; ++j)
                acc[i][j] = __builtin_amdgcn_wmma_f32_16x16x32_f16(
                    false, af[i], false, bf[j], (short)0, acc[i][j], false, false);
    }

    // D layout: lanes 0-15 -> n = lane, m = vgpr; lanes 16-31 -> m = vgpr + 8.
    const int ncol = lane & 15;
    const int moff = (lane >> 4) * 8;
#pragma unroll
    for (int j = 0; j < 2; ++j) {
        const int n = (nt0 + j) * 16 + ncol;
        const float bs = bias[n];
#pragma unroll
        for (int i = 0; i < 4; ++i) {
#pragma unroll
            for (int r = 0; r < 8; ++r) {
                const int m = (mt0 + i) * 16 + moff + r;
                const float val = acc[i][j][r] + bs;
                if (F16OUT) ((_Float16*)Cout)[(size_t)m * N + n] = (_Float16)val;
                else        ((float*)Cout)[(size_t)m * N + n]    = val;
            }
        }
    }
}

// ---------------------------------------------------------------------------
// Banded causal attention, window NWIN. qkv16: [B*T, 3*D] f16 (Q|K|V blocks).
// One thread per (b, h, t). Writes f16 [B*T, D] in (b,t,h*64+d) layout.
// ---------------------------------------------------------------------------
__global__ void attn_kernel(const _Float16* __restrict__ qkv,
                            _Float16* __restrict__ out) {
    int g = blockIdx.x * blockDim.x + threadIdx.x;       // (b*H + h)*T + t
    if (g >= NB * NH * NT) return;
    const int t  = g % NT;
    const int bh = g / NT;
    const int h  = bh % NH;
    const int b  = bh / NH;
    const size_t rowbase = ((size_t)b * NT + t) * (3 * ND);
    const size_t hoff = (size_t)h * NDH;

    float q[NDH];
    {
        const v8h* qp = (const v8h*)(qkv + rowbase + hoff);
#pragma unroll
        for (int c = 0; c < 8; ++c) {
            v8h x = qp[c];
#pragma unroll
            for (int e = 0; e < 8; ++e) q[c * 8 + e] = (float)x[e];
        }
    }

    const int nj = (t + 1 < NWIN) ? (t + 1) : NWIN;
    float s[NWIN];
    float mx = -3.0e38f;
    for (int j = 0; j < nj; ++j) {
        const v8h* kp = (const v8h*)(qkv + rowbase - (size_t)j * (3 * ND) + ND + hoff);
        float a = 0.f;
#pragma unroll
        for (int c = 0; c < 8; ++c) {
            v8h x = kp[c];
#pragma unroll
            for (int e = 0; e < 8; ++e) a += q[c * 8 + e] * (float)x[e];
        }
        a *= 0.125f;                                     // 1/sqrt(64)
        s[j] = a;
        mx = fmaxf(mx, a);
    }
    float sum = 0.f;
    for (int j = 0; j < nj; ++j) { s[j] = expf(s[j] - mx); sum += s[j]; }
    const float inv = 1.0f / sum;

    float o[NDH];
#pragma unroll
    for (int d = 0; d < NDH; ++d) o[d] = 0.f;
    for (int j = 0; j < nj; ++j) {
        const float p = s[j] * inv;
        const v8h* vp = (const v8h*)(qkv + rowbase - (size_t)j * (3 * ND) + 2 * ND + hoff);
#pragma unroll
        for (int c = 0; c < 8; ++c) {
            v8h x = vp[c];
#pragma unroll
            for (int e = 0; e < 8; ++e) o[c * 8 + e] += p * (float)x[e];
        }
    }
    v8h* op = (v8h*)(out + ((size_t)b * NT + t) * ND + hoff);
#pragma unroll
    for (int c = 0; c < 8; ++c) {
        v8h x;
#pragma unroll
        for (int e = 0; e < 8; ++e) x[e] = (_Float16)o[c * 8 + e];
        op[c] = x;
    }
}

// ---------------------------------------------------------------------------
// LayerNorm over last dim (512): out = LN(a + b) * w + bias.
// One wave (32 lanes) per row; wave32 shfl_xor reduction.
// ---------------------------------------------------------------------------
__global__ void ln_kernel(const float* __restrict__ a, const float* __restrict__ b,
                          const float* __restrict__ w, const float* __restrict__ bias,
                          float* __restrict__ out, int M) {
    const int wave = threadIdx.x >> 5;
    const int lane = threadIdx.x & 31;
    const int row  = blockIdx.x * 8 + wave;
    if (row >= M) return;
    const float* pa = a + (size_t)row * ND;
    const float* pb = b + (size_t)row * ND;
    float x[16];
    float s = 0.f, sq = 0.f;
#pragma unroll
    for (int k = 0; k < 16; ++k) {
        const int c = lane + k * 32;
        const float v = pa[c] + pb[c];
        x[k] = v; s += v; sq += v * v;
    }
#pragma unroll
    for (int off = 16; off >= 1; off >>= 1) {
        s  += __shfl_xor(s,  off, 32);
        sq += __shfl_xor(sq, off, 32);
    }
    const float mean = s * (1.0f / ND);
    const float var  = sq * (1.0f / ND) - mean * mean;
    const float rstd = rsqrtf(var + LN_EPS);
    float* po = out + (size_t)row * ND;
#pragma unroll
    for (int k = 0; k < 16; ++k) {
        const int c = lane + k * 32;
        po[c] = (x[k] - mean) * rstd * w[c] + bias[c];
    }
}

// ---------------------------------------------------------------------------
extern "C" void kernel_launch(void* const* d_in, const int* in_sizes, int n_in,
                              void* d_out, int out_size, void* d_ws, size_t ws_size,
                              hipStream_t stream) {
    const float* z    = (const float*)d_in[0];
    const float* Wq   = (const float*)d_in[1];
    const float* bq   = (const float*)d_in[2];
    const float* Wk   = (const float*)d_in[3];
    const float* bk   = (const float*)d_in[4];
    const float* Wv   = (const float*)d_in[5];
    const float* bv   = (const float*)d_in[6];
    const float* Wo   = (const float*)d_in[7];
    const float* bo   = (const float*)d_in[8];
    const float* ln1w = (const float*)d_in[9];
    const float* ln1b = (const float*)d_in[10];
    const float* W1   = (const float*)d_in[11];
    const float* b1   = (const float*)d_in[12];
    const float* W2   = (const float*)d_in[13];
    const float* b2   = (const float*)d_in[14];
    const float* ln2w = (const float*)d_in[15];
    const float* ln2b = (const float*)d_in[16];
    float* outp = (float*)d_out;

    const int M = NB * NT;                 // 8192 rows

    // Workspace carve-up (lifetime-based reuse).
    char* ws = (char*)d_ws;
    size_t off = 0;
    auto carve = [&](size_t bytes) { char* p = ws + off; off += (bytes + 255) & ~(size_t)255; return p; };
    _Float16* wqkv_pk = (_Float16*)carve((size_t)ND * (3 * ND) * 2);   // 1.5 MB
    _Float16* wo_pk   = (_Float16*)carve((size_t)ND * ND * 2);
    _Float16* w1_pk   = (_Float16*)carve((size_t)ND * NFF * 2);
    _Float16* w2_pk   = (_Float16*)carve((size_t)NFF * ND * 2);
    float*    bias_qkv= (float*)carve((size_t)(3 * ND) * 4);
    char*     bufA    = carve((size_t)M * NFF * 2);   // qkv16 (25MB) then h1_f16 (33.5MB)
    char*     bufB    = carve((size_t)M * ND * 2);    // z_pk -> attn_pk -> x1_pk
    char*     bufC    = carve((size_t)M * NFF * 2);   // attn16 then gelu_pk
    float*    bufD    = (float*)carve((size_t)M * ND * 4); // y1 then y2
    float*    bufE    = (float*)carve((size_t)M * ND * 4); // x1
    (void)ws_size;

    const int BS = 256;
    // 1) Weight packing + bias concat.
    concat_bias_kernel<<<2, BS, 0, stream>>>(bq, bk, bv, bias_qkv);
    {
        const int blocks = (ND * ND / 2 + BS - 1) / BS;  // 512 blocks per 512x512
        pack_b_kernel<<<blocks, BS, 0, stream>>>(Wq, wqkv_pk,                 ND, ND);
        pack_b_kernel<<<blocks, BS, 0, stream>>>(Wk, wqkv_pk + (size_t)ND*ND, ND, ND);
        pack_b_kernel<<<blocks, BS, 0, stream>>>(Wv, wqkv_pk + (size_t)2*ND*ND, ND, ND);
        pack_b_kernel<<<blocks, BS, 0, stream>>>(Wo, wo_pk, ND, ND);
        const int blocks_ff = (ND * NFF / 2 + BS - 1) / BS;
        pack_b_kernel<<<blocks_ff, BS, 0, stream>>>(W1, w1_pk, ND, NFF);
        pack_b_kernel<<<blocks_ff, BS, 0, stream>>>(W2, w2_pk, NFF, ND);
    }
    // 2) Pack z as A fragments.
    pack_a_kernel<float, false><<<(M * ND / 2) / BS, BS, 0, stream>>>(z, (_Float16*)bufB, M, ND);
    // 3) QKV GEMM: [8192,512] x [512,1536] -> f16 qkv.  (M/64)*(N/32)/8 blocks.
    gemm_wmma_kernel<true><<<(M / 64) * (3 * ND / 32) / 8, BS, 0, stream>>>(
        (const v16h*)bufB, (const v16h*)wqkv_pk, bias_qkv, bufA, M, 3 * ND, ND);
    // 4) Banded causal attention.
    attn_kernel<<<(NB * NH * NT) / BS, BS, 0, stream>>>((const _Float16*)bufA, (_Float16*)bufC);
    // 5) Pack attention output.
    pack_a_kernel<_Float16, false><<<(M * ND / 2) / BS, BS, 0, stream>>>(
        (const _Float16*)bufC, (_Float16*)bufB, M, ND);
    // 6) Output projection: y1 = attn @ Wo + bo (f32).
    gemm_wmma_kernel<false><<<(M / 64) * (ND / 32) / 8, BS, 0, stream>>>(
        (const v16h*)bufB, (const v16h*)wo_pk, bo, bufD, M, ND, ND);
    // 7) x1 = LN(z + y1).
    ln_kernel<<<M / 8, BS, 0, stream>>>(z, bufD, ln1w, ln1b, bufE, M);
    // 8) Pack x1.
    pack_a_kernel<float, false><<<(M * ND / 2) / BS, BS, 0, stream>>>(bufE, (_Float16*)bufB, M, ND);
    // 9) FFN up: h1 = x1 @ W1 + b1 (f16 out).
    gemm_wmma_kernel<true><<<(M / 64) * (NFF / 32) / 8, BS, 0, stream>>>(
        (const v16h*)bufB, (const v16h*)w1_pk, b1, bufA, M, NFF, ND);
    // 10) Exact GELU fused into A-fragment pack.
    pack_a_kernel<_Float16, true><<<(M * NFF / 2) / BS, BS, 0, stream>>>(
        (const _Float16*)bufA, (_Float16*)bufC, M, NFF);
    // 11) FFN down: y2 = gelu(h1) @ W2 + b2 (f32).
    gemm_wmma_kernel<false><<<(M / 64) * (ND / 32) / 8, BS, 0, stream>>>(
        (const v16h*)bufC, (const v16h*)w2_pk, b2, bufD, M, ND, NFF);
    // 12) out = LN(x1 + y2).
    ln_kernel<<<M / 8, BS, 0, stream>>>(bufE, bufD, ln2w, ln2b, outp, M);
}
